// AgentPolicy_57930518889164
// MI455X (gfx1250) — matrix operand order repaired and checked by reading, not compiled, not run on hardware
//
#include <hip/hip_runtime.h>
#include <hip/hip_bf16.h>

typedef _Float16 half_t;
typedef __attribute__((ext_vector_type(16))) _Float16 v16h;
typedef __attribute__((ext_vector_type(8)))  _Float16 v8h;
typedef __attribute__((ext_vector_type(4)))  _Float16 v4h;
typedef __attribute__((ext_vector_type(2)))  _Float16 v2h;
typedef __attribute__((ext_vector_type(8)))  float    v8f;
typedef __attribute__((ext_vector_type(4)))  float    v4f;

#define B_SZ 4096
#define N_AG 64
#define FIN  256
#define D1   512
#define D2   128
#define NEGV (-10000000.0f)

// LDS layout (halves strides padded to dodge bank conflicts on A-frag loads)
#define SA 264              // A row stride in halves (528 B = 132 dw -> +4 banks/row)
#define SY 520              // Y row stride in halves
#define OFF_A     0
#define OFF_Y     (80 * SA * 2)              // 42240
#define OFF_ALPHA (OFF_Y + 80 * SY * 2)      // 125440
#define OFF_PART  (OFF_ALPHA + 64 * 4)       // 125696
#define OFF_OUT   (OFF_PART + 256 * 4)       // 126720
#define OFF_LOG   (OFF_OUT + 1024 * 4)       // 130816
#define OFF_RED   (OFF_LOG + 256 * 4)        // 131840
#define SMEM_BYTES (OFF_RED + 64)            // 131904 (< 320KB/WGP, 2 WGs/WGP)

// W1 [FIN][D1] f32 -> W1t [D1][FIN] f16 (transposed so B-fragments are contiguous)
__global__ void convert_w1(const float* __restrict__ W1, half_t* __restrict__ W1t) {
    int idx = blockIdx.x * 256 + threadIdx.x;   // 0 .. D1*FIN-1
    int d = idx >> 8;                           // / FIN
    int k = idx & 255;
    W1t[idx] = (half_t)W1[k * D1 + d];
}

__device__ __forceinline__ void cvt_store4(half_t* dst, v4f v) {
    v4h h;
    h[0] = (half_t)v[0]; h[1] = (half_t)v[1];
    h[2] = (half_t)v[2]; h[3] = (half_t)v[3];
    *(v4h*)dst = h;                              // ds_store_b64
}

__global__ __launch_bounds__(256) void agent_policy(
    const float* __restrict__ obs_x, const float* __restrict__ others,
    const int* __restrict__ amask, const half_t* __restrict__ W1t,
    const float* __restrict__ b1, const float* __restrict__ W2,
    const float* __restrict__ b2, float* __restrict__ outp)
{
    extern __shared__ char smem[];
    half_t* As    = (half_t*)(smem + OFF_A);      // [80][SA] f16 : rows 0-63 others, 64 obs_x, 65-79 zero
    half_t* Ys    = (half_t*)(smem + OFF_Y);      // [80][SY] f16 : GEMM output (+b1)
    float*  alpha = (float*) (smem + OFF_ALPHA);  // [64]
    float*  part  = (float*) (smem + OFF_PART);   // [64][4]
    float*  outv  = (float*) (smem + OFF_OUT);    // [1024] = concat(x, c)
    float*  lg    = (float*) (smem + OFF_LOG);    // [256] fc2 partials / logits
    float*  red   = (float*) (smem + OFF_RED);

    const int b = blockIdx.x;
    const int t = threadIdx.x;

    // ---- stage A into LDS as f16 (vectorized: b128 loads, b64 stores) ----
    {
        const v4f* ob4 = (const v4f*)(others + (size_t)b * (N_AG * FIN));
        // 64*256/4 = 4096 float4 chunks, 16 per thread; 64 chunks per row
        for (int i = t; i < (N_AG * FIN) / 4; i += 256) {
            const int n = i >> 6;            // row
            const int k = (i & 63) * 4;      // col
            cvt_store4(&As[n * SA + k], ob4[i]);
        }
        // row 64 = obs_x (256 floats = 64 chunks, threads 0..63)
        if (t < 64) {
            const v4f* ox4 = (const v4f*)(obs_x + (size_t)b * FIN);
            cvt_store4(&As[64 * SA + t * 4], ox4[t]);
        }
        // rows 65..79 = 0  (15*64 = 960 chunks)
        const v4h z = {};
        for (int i = t; i < 15 * 64; i += 256) {
            const int n = 65 + (i >> 6);
            const int k = (i & 63) * 4;
            *(v4h*)&As[n * SA + k] = z;
        }
    }
    __syncthreads();

    // ---- GEMM: [80x256] @ W1t^T -> Y [80x512], f16 WMMA, f32 accum ----
    const int w    = t >> 5;
    const int lane = t & 31;
    const int lhi  = (lane >> 4) & 1;
    const int l15  = lane & 15;

    for (int nt = w * 4; nt < w * 4 + 4; ++nt) {
        // B fragments for this ntile: lane = column nt*16+l15, 16 contiguous K halves
        v16h bfr[8];
        const half_t* bp = W1t + (size_t)(nt * 16 + l15) * FIN;
        for (int kt = 0; kt < 8; ++kt)
            bfr[kt] = *(const v16h*)(bp + kt * 32 + lhi * 16);
        const float bias = b1[nt * 16 + l15];

        for (int mt = 0; mt < 5; ++mt) {
            v8f acc = {};
            const half_t* arow = As + (mt * 16 + l15) * SA;
            for (int kt = 0; kt < 8; ++kt) {
                const int kb = kt * 32 + lhi * 16;
                v8h lo = *(const v8h*)(arow + kb);
                v8h hi = *(const v8h*)(arow + kb + 8);
                v16h a;
                for (int i = 0; i < 8; ++i) { a[i] = lo[i]; a[i + 8] = hi[i]; }
                acc = __builtin_amdgcn_wmma_f32_16x16x32_f16(
                        false, a, false, bfr[kt], (short)0, acc, false, false);
            }
            const int col     = nt * 16 + l15;
            const int rowbase = mt * 16 + lhi * 8;
            for (int v = 0; v < 8; ++v)
                Ys[(rowbase + v) * SY + col] = (half_t)(acc[v] + bias);
        }
    }
    __syncthreads();

    // ---- alpha[n] = (y_n . x) / sqrt(D1) ----
    {
        const int n = t >> 2, q = t & 3;
        const v2h* yr = (const v2h*)(Ys + n * SY + q * 128);
        const v2h* xr = (const v2h*)(Ys + 64 * SY + q * 128);
        float s = 0.0f;
        for (int d = 0; d < 64; ++d) {       // 64 pairs = 128 halves
            v2h yv = yr[d], xv = xr[d];
            s += (float)yv[0] * (float)xv[0] + (float)yv[1] * (float)xv[1];
        }
        part[t] = s;
    }
    __syncthreads();
    if (t < 64)
        alpha[t] = (part[t*4] + part[t*4+1] + part[t*4+2] + part[t*4+3])
                   * 0.044194173824159216f;   // 1/sqrt(512)
    __syncthreads();
    if (t == 0) {   // softmax over 64 (trivial serial cost)
        float mx = alpha[0];
        for (int n = 1; n < 64; ++n) mx = fmaxf(mx, alpha[n]);
        float sum = 0.0f;
        for (int n = 0; n < 64; ++n) { float e = expf(alpha[n] - mx); alpha[n] = e; sum += e; }
        float inv = 1.0f / sum;
        for (int n = 0; n < 64; ++n) alpha[n] *= inv;
    }
    __syncthreads();

    // ---- out = concat(x, c) with c[d] = sum_n beta[n] * y[n][d] (v2h reads) ----
    {
        const int d = t * 2;                 // two consecutive d's per thread
        v2h xv = *(const v2h*)(Ys + 64 * SY + d);
        outv[d]     = (float)xv[0];
        outv[d + 1] = (float)xv[1];
        float c0 = 0.0f, c1 = 0.0f;
        for (int n = 0; n < 64; ++n) {
            const float bn = alpha[n];
            v2h yv = *(const v2h*)(Ys + n * SY + d);
            c0 += bn * (float)yv[0];
            c1 += bn * (float)yv[1];
        }
        outv[D1 + d]     = c0;
        outv[D1 + d + 1] = c1;
    }
    __syncthreads();

    // ---- fc2: logits = out @ W2 + b2 (2 threads per column) ----
    {
        const int col = t & 127;
        const int kh  = t >> 7;
        float acc = (kh == 0) ? b2[col] : 0.0f;
        const float* w2p = W2 + (size_t)(kh * 512) * D2 + col;
        const float* op  = outv + kh * 512;
        for (int k = 0; k < 512; ++k)
            acc += op[k] * w2p[(size_t)k * D2];
        lg[kh * 128 + col] = acc;
    }
    __syncthreads();
    if (t < 128) lg[t] = lg[t] + lg[128 + t];
    __syncthreads();
    if (t == 0) {   // softmax stats over 128
        float mx = lg[0];
        for (int i = 1; i < 128; ++i) mx = fmaxf(mx, lg[i]);
        float sum = 0.0f;
        for (int i = 0; i < 128; ++i) sum += expf(lg[i] - mx);
        red[0] = mx; red[1] = 1.0f / sum;
    }
    __syncthreads();
    if (t < 128) {
        float p = expf(lg[t] - red[0]) * red[1];
        float m = (float)amask[(size_t)b * D2 + t];
        outp[(size_t)b * D2 + t] = p + NEGV * (1.0f - m);
    }
}

extern "C" void kernel_launch(void* const* d_in, const int* in_sizes, int n_in,
                              void* d_out, int out_size, void* d_ws, size_t ws_size,
                              hipStream_t stream) {
    const float* obs_x  = (const float*)d_in[0];
    const float* others = (const float*)d_in[1];
    const int*   amask  = (const int*)  d_in[2];
    const float* W1     = (const float*)d_in[3];
    const float* b1     = (const float*)d_in[4];
    const float* W2     = (const float*)d_in[5];
    const float* b2     = (const float*)d_in[6];
    // d_in[7..10] = W3, b3, W4, b4: value head is discarded by the reference -> skipped.

    half_t* W1t = (half_t*)d_ws;   // 512*256*2 = 256 KB scratch

    (void)hipFuncSetAttribute((const void*)agent_policy,
                              hipFuncAttributeMaxDynamicSharedMemorySize, SMEM_BYTES);

    convert_w1<<<(D1 * FIN) / 256, 256, 0, stream>>>(W1, W1t);
    agent_policy<<<B_SZ, 256, SMEM_BYTES, stream>>>(obs_x, others, amask, W1t,
                                                    b1, W2, b2, (float*)d_out);
}
